// GCN_WalkPooling_29867202577016
// MI455X (gfx1250) — matrix operand order repaired and checked by reading, not compiled, not run on hardware
//
#include <hip/hip_runtime.h>
#include <hip/hip_bf16.h>
#include <stdint.h>

typedef __attribute__((ext_vector_type(16))) __bf16   v16bf;
typedef __attribute__((ext_vector_type(8)))  float    v8f;
typedef __attribute__((ext_vector_type(8)))  uint32_t v8u;

namespace {
constexpr int BB    = 256;
constexpr int NP    = 128;
constexpr int NN    = BB * NP;        // 32768 nodes
constexpr int DEGC  = 8;
constexpr int EREAL = NN * DEGC;      // 262144
constexpr int EE    = EREAL + 2 * BB; // 262656 edges
constexpr int HH    = 32;
constexpr int HEADS = 2;
constexpr int WALK  = 7;
constexpr float LRELU_S = 0.2f;
constexpr size_t DSZ = (size_t)BB * HEADS * NP * NP;  // dense walk matrix elems per variant
}

// ---------- helpers ----------
static __device__ __forceinline__ unsigned short f2bf(float f) {
  uint32_t u = __float_as_uint(f);
  uint32_t r = (u + 0x7FFFu + ((u >> 16) & 1u)) >> 16;   // RNE
  return (unsigned short)r;
}
static __device__ __forceinline__ float bf2f(unsigned short h) {
  return __uint_as_float(((uint32_t)h) << 16);
}
// order-preserving float->uint for atomicMax
static __device__ __forceinline__ uint32_t fenc(float f) {
  uint32_t u = __float_as_uint(f);
  return (u & 0x80000000u) ? ~u : (u | 0x80000000u);
}
static __device__ __forceinline__ float fdec(uint32_t k) {
  return __uint_as_float((k & 0x80000000u) ? (k & 0x7FFFFFFFu) : ~k);
}
static __device__ __forceinline__ v16bf pack_frag(uint4 lo, uint4 hi) {
  v8u t;
  t[0] = lo.x; t[1] = lo.y; t[2] = lo.z; t[3] = lo.w;
  t[4] = hi.x; t[5] = hi.y; t[6] = hi.z; t[7] = hi.w;
  v16bf r; __builtin_memcpy(&r, &t, sizeof(r)); return r;
}

// ---------- generic fills ----------
__global__ void k_fillf(float* p, size_t n, float v) {
  size_t i = (size_t)blockIdx.x * blockDim.x + threadIdx.x;
  if (i < n) p[i] = v;
}
__global__ void k_fillu(uint32_t* p, size_t n, uint32_t v) {
  size_t i = (size_t)blockIdx.x * blockDim.x + threadIdx.x;
  if (i < n) p[i] = v;
}

// ---------- degree / embedding ----------
__global__ void k_deg(const int* __restrict__ col, float* __restrict__ deg) {
  int e = blockIdx.x * blockDim.x + threadIdx.x;
  if (e < EE) atomicAdd(&deg[col[e]], 1.0f);
}
__global__ void k_dinv(float* __restrict__ deg) {
  int n = blockIdx.x * blockDim.x + threadIdx.x;
  if (n < NN) deg[n] = rsqrtf(deg[n]);
}
__global__ void k_embed(const float* __restrict__ zt, const int* __restrict__ z,
                        float* __restrict__ x) {
  int i = blockIdx.x * blockDim.x + threadIdx.x;
  if (i < NN * HH) { int n = i >> 5, h = i & 31; x[i] = zt[z[n] * HH + h]; }
}

// ---------- GCN conv ----------
__global__ void k_xw(const float* __restrict__ x, const float* __restrict__ W,
                     float* __restrict__ xw) {
  int i = blockIdx.x * blockDim.x + threadIdx.x;
  if (i >= NN * HH) return;
  int n = i >> 5, j = i & 31;
  float a = 0.f;
  const float* xr = x + n * HH;
#pragma unroll 8
  for (int k = 0; k < HH; ++k) a += xr[k] * W[k * HH + j];
  xw[i] = a;
}
__global__ void k_selfinit(const float* __restrict__ xw, const float* __restrict__ dinv,
                           const float* __restrict__ bias, float* __restrict__ out) {
  int i = blockIdx.x * blockDim.x + threadIdx.x;
  if (i >= NN * HH) return;
  int n = i >> 5, j = i & 31;
  out[i] = dinv[n] * dinv[n] * xw[i] + bias[j];
}
__global__ void k_scatter_conv(const int* __restrict__ row, const int* __restrict__ col,
                               const float* __restrict__ dinv, const float* __restrict__ xw,
                               float* __restrict__ out) {
  size_t i = (size_t)blockIdx.x * blockDim.x + threadIdx.x;
  if (i >= (size_t)EE * HH) return;
  int e = (int)(i >> 5), j = (int)(i & 31);
  int r = row[e], c = col[e];
  atomicAdd(&out[c * HH + j], dinv[r] * dinv[c] * xw[r * HH + j]);
}
__global__ void k_act(const float* __restrict__ in, float* __restrict__ x, int relu) {
  int i = blockIdx.x * blockDim.x + threadIdx.x;
  if (i < NN * HH) { float v = in[i]; x[i] = (relu && v < 0.f) ? 0.f : v; }
}

// ---------- attention projections ----------
__global__ void k_qk1(const float* __restrict__ x,
                      const float* __restrict__ k1W, const float* __restrict__ k1b,
                      const float* __restrict__ q1W, const float* __restrict__ q1b,
                      float* __restrict__ qa, float* __restrict__ ka) {
  int i = blockIdx.x * blockDim.x + threadIdx.x;
  if (i >= NN * HH) return;
  int n = i >> 5, j = i & 31;
  const float* xr = x + n * HH;
  float aq = k1b[j], ak = q1b[j];
#pragma unroll 8
  for (int k = 0; k < HH; ++k) { float xv = xr[k]; aq += xv * k1W[k * HH + j]; ak += xv * q1W[k * HH + j]; }
  qa[i] = aq > 0.f ? aq : LRELU_S * aq;
  ka[i] = ak > 0.f ? ak : LRELU_S * ak;
}
__global__ void k_qk2(const float* __restrict__ qa, const float* __restrict__ ka,
                      const float* __restrict__ k2W, const float* __restrict__ k2b,
                      const float* __restrict__ q2W, const float* __restrict__ q2b,
                      float* __restrict__ qb, float* __restrict__ kb) {
  int i = blockIdx.x * blockDim.x + threadIdx.x;
  if (i >= NN * 64) return;
  int n = i >> 6, j = i & 63;
  const float* qr = qa + n * HH;
  const float* kr = ka + n * HH;
  float aq = k2b[j], ak = q2b[j];
#pragma unroll 8
  for (int k = 0; k < HH; ++k) { aq += qr[k] * k2W[k * 64 + j]; ak += kr[k] * q2W[k * 64 + j]; }
  qb[i] = aq; kb[i] = ak;
}

// ---------- edge scores + segment softmax ----------
__global__ void k_edge_w(const int* __restrict__ row, const int* __restrict__ col,
                         const float* __restrict__ qb, const float* __restrict__ kb,
                         float* __restrict__ w, uint32_t* __restrict__ wkey) {
  int i = blockIdx.x * blockDim.x + threadIdx.x;
  if (i >= EE * HEADS) return;
  int e = i >> 1, h = i & 1;
  const float* qr = qb + (size_t)row[e] * 64 + h * 32;
  const float* kr = kb + (size_t)col[e] * 64 + h * 32;
  float s = 0.f;
#pragma unroll 8
  for (int k = 0; k < 32; ++k) s += qr[k] * kr[k];
  s *= 0.17677669529663687f;  // 1/sqrt(32)
  w[i] = s;
  atomicMax(&wkey[col[e] * HEADS + h], fenc(s));
}
__global__ void k_edge_ew(const int* __restrict__ col, const float* __restrict__ w,
                          const uint32_t* __restrict__ wkey, float* __restrict__ ew,
                          float* __restrict__ sump, float* __restrict__ summ) {
  int i = blockIdx.x * blockDim.x + threadIdx.x;
  if (i >= EE * HEADS) return;
  int e = i >> 1, h = i & 1;
  float v = __expf(w[i] - fdec(wkey[col[e] * HEADS + h]));
  ew[i] = v;
  atomicAdd(&sump[col[e] * HEADS + h], v);
  if (e < EREAL) atomicAdd(&summ[col[e] * HEADS + h], v);
}
__global__ void k_omega(const float* __restrict__ w, float* __restrict__ om) {
  int i = blockIdx.x * blockDim.x + threadIdx.x;
  if (i >= BB * HEADS) return;
  int b = i >> 1, h = i & 1;
  float w0 = w[(size_t)(EREAL + 2 * b) * HEADS + h];
  float w1 = w[(size_t)(EREAL + 2 * b + 1) * HEADS + h];
  om[i] = 1.f / (1.f + __expf(-w0)) + 1.f / (1.f + __expf(-w1));
}

// ---------- dense walk matrices ----------
__global__ void k_scatter_dense(const int* __restrict__ row, const int* __restrict__ col,
                                const float* __restrict__ ew, const float* __restrict__ sump,
                                const float* __restrict__ summ,
                                float* __restrict__ dp, float* __restrict__ dm) {
  int i = blockIdx.x * blockDim.x + threadIdx.x;
  if (i >= EE * HEADS) return;
  int e = i >> 1, h = i & 1;
  int cg = col[e], rg = row[e];
  int g = cg >> 7, c = cg & 127, r = rg & 127;
  size_t base = (((size_t)g * HEADS + h) * NP + c) * NP + r;
  float v = ew[i] / (sump[cg * HEADS + h] + 1e-16f);
  atomicAdd(&dp[base], v);
  if (e < EREAL) {
    float vm = ew[i] / (summ[cg * HEADS + h] + 1e-16f);
    atomicAdd(&dm[base], vm);
  }
}
__global__ void k_tobf(const float* __restrict__ src, unsigned short* __restrict__ dst, size_t n) {
  size_t i = (size_t)blockIdx.x * blockDim.x + threadIdx.x;
  if (i < n) dst[i] = f2bf(src[i]);
}

// ---------- walk pooling: 8 chained 128x128x128 bf16 WMMA GEMMs in LDS ----------
extern __shared__ unsigned short walk_smem[];

__global__ __launch_bounds__(256)
void k_walk(const unsigned short* __restrict__ Wp, const unsigned short* __restrict__ Wm,
            const int* __restrict__ node_i, const int* __restrict__ node_j,
            float* __restrict__ tr, float* __restrict__ nl, float* __restrict__ ll) {
  // plain scalar LDS pointers only (no aggregates -> no addrspacecast initializers)
  unsigned short* As = walk_smem;                // W_dense [c][r], bf16, 32KB
  unsigned short* X0 = walk_smem + NP * NP;      // state ping [n][r]
  unsigned short* X1 = walk_smem + 2 * NP * NP;  // state pong [n][r]

  const int tid = threadIdx.x;
  const int v = blockIdx.x & 1, h = (blockIdx.x >> 1) & 1, g = blockIdx.x >> 2;
  const unsigned short* Wsrc = (v == 0 ? Wp : Wm) + ((size_t)g * HEADS + h) * NP * NP;

  // pull the 32KB dense matrix toward the caches (global_prefetch_b8)
  __builtin_prefetch(Wsrc + (size_t)tid * 64, 0, 3);

  { // load dense walk matrix (32KB) with 128-bit vmem loads
    const uint4* s = (const uint4*)Wsrc;
    uint4* d = (uint4*)As;
    for (int i = tid; i < NP * NP / 8; i += 256) d[i] = s[i];
  }
  for (int i = tid; i < NP * NP; i += 256)       // x0 = I (transposed storage)
    X0[i] = ((i >> 7) == (i & 127)) ? (unsigned short)0x3F80 : (unsigned short)0;
  __syncthreads();

  const int lane = tid & 31, wave = tid >> 5;
  const int half = lane >> 4, l15 = lane & 15;
  const int iloc = node_i[g] & 127, jloc = node_j[g] & 127;
  const int mt = wave;  // each wave owns one 16-row output strip
  int cur = 0;

  // A fragments: constant across steps/tiles -> load once, stay in VGPRs.
  // A 16x32 bf16 layout: lane l15 -> M; dword pairs contiguous in K:
  //   dwords 0..3 at K-dword kt*16 + half*4 + i ; dwords 4..7 at +8.
  v16bf afrag[4];
#pragma unroll
  for (int kt = 0; kt < 4; ++kt) {
    const uint4* Arow = (const uint4*)(As + (mt * 16 + l15) * NP);
    uint4 lo = Arow[kt * 4 + half];
    uint4 hi = Arow[kt * 4 + 2 + half];
    afrag[kt] = pack_frag(lo, hi);
  }

  for (int step = 0; step < 8; ++step) {
    unsigned short* Xc = (cur == 0) ? X0 : X1;
    unsigned short* Xn = (cur == 0) ? X1 : X0;
    // two output tiles in flight -> independent accumulators break the
    // bf16 WMMA->WMMA RAW hazard (no v_nop padding needed)
    for (int ntp = 0; ntp < 4; ++ntp) {
      const int nt0 = 2 * ntp, nt1 = 2 * ntp + 1;
      v8f acc0 = {0.f, 0.f, 0.f, 0.f, 0.f, 0.f, 0.f, 0.f};
      v8f acc1 = {0.f, 0.f, 0.f, 0.f, 0.f, 0.f, 0.f, 0.f};
      const uint4* B0 = (const uint4*)(Xc + (nt0 * 16 + l15) * NP);
      const uint4* B1 = (const uint4*)(Xc + (nt1 * 16 + l15) * NP);
#pragma unroll
      for (int kt = 0; kt < 4; ++kt) {
        // B 32x16 bf16 layout: lane l15 -> N; same K order as A:
        // dwords 0..3 at K-dword kt*16 + half*4 + i ; dwords 4..7 at +8.
        const int o = kt * 4 + half;
        v16bf b0 = pack_frag(B0[o], B0[o + 2]);
        v16bf b1 = pack_frag(B1[o], B1[o + 2]);
        acc0 = __builtin_amdgcn_wmma_f32_16x16x32_bf16(
            false, afrag[kt], false, b0, (short)0, acc0, false, false);
        acc1 = __builtin_amdgcn_wmma_f32_16x16x32_bf16(
            false, afrag[kt], false, b1, (short)0, acc1, false, false);
      }
      // C layout: lane l15 -> N; dword j -> M = mt*16 + j + half*8.
      // The 8 bf16 results are contiguous rows in the transposed state ->
      // one 16B packed store (ds_store_b128) per tile.
      uint4 s0, s1;
      s0.x = (uint32_t)f2bf(acc0[0]) | ((uint32_t)f2bf(acc0[1]) << 16);
      s0.y = (uint32_t)f2bf(acc0[2]) | ((uint32_t)f2bf(acc0[3]) << 16);
      s0.z = (uint32_t)f2bf(acc0[4]) | ((uint32_t)f2bf(acc0[5]) << 16);
      s0.w = (uint32_t)f2bf(acc0[6]) | ((uint32_t)f2bf(acc0[7]) << 16);
      s1.x = (uint32_t)f2bf(acc1[0]) | ((uint32_t)f2bf(acc1[1]) << 16);
      s1.y = (uint32_t)f2bf(acc1[2]) | ((uint32_t)f2bf(acc1[3]) << 16);
      s1.z = (uint32_t)f2bf(acc1[4]) | ((uint32_t)f2bf(acc1[5]) << 16);
      s1.w = (uint32_t)f2bf(acc1[6]) | ((uint32_t)f2bf(acc1[7]) << 16);
      *(uint4*)(Xn + (nt0 * 16 + l15) * NP + mt * 16 + half * 8) = s0;
      *(uint4*)(Xn + (nt1 * 16 + l15) * NP + mt * 16 + half * 8) = s1;
    }
    __syncthreads();
    if (step >= 1 && tid == 0) {
      float t = 0.f;
      for (int d = 0; d < NP; ++d) t += bf2f(Xn[d * NP + d]);   // trace
      float nlv = bf2f(Xn[iloc * NP + iloc]) + bf2f(Xn[jloc * NP + jloc]);
      float llv = bf2f(Xn[jloc * NP + iloc]) + bf2f(Xn[iloc * NP + jloc]);
      const size_t o = ((size_t)(v * WALK + (step - 1)) * HEADS + h) * BB + g;
      tr[o] = t; nl[o] = nlv; ll[o] = llv;
    }
    cur ^= 1;
  }
}

// ---------- final MLP ----------
__global__ void k_mlp(const float* __restrict__ tr, const float* __restrict__ nl,
                      const float* __restrict__ ll, const float* __restrict__ om,
                      const float* __restrict__ W1, const float* __restrict__ b1,
                      const float* __restrict__ W2, const float* __restrict__ b2,
                      float* __restrict__ out) {
  __shared__ float feat[72];
  __shared__ float red[32];
  const int b = blockIdx.x, j = threadIdx.x;
  for (int f = j; f < 72; f += 32) {
    float v; int q, hh, ss;
    if (f < 14)      { hh = f / 7; ss = f % 7;
                       v = tr[((size_t)ss * HEADS + hh) * BB + b]
                         - tr[((size_t)(WALK + ss) * HEADS + hh) * BB + b]; }
    else if (f < 16)   v = om[b * 2 + (f - 14)];
    else if (f < 30) { q = f - 16; hh = q / 7; ss = q % 7;
                       v = nl[((size_t)ss * HEADS + hh) * BB + b]; }
    else if (f < 44) { q = f - 30; hh = q / 7; ss = q % 7;
                       v = nl[((size_t)(WALK + ss) * HEADS + hh) * BB + b]; }
    else if (f < 58) { q = f - 44; hh = q / 7; ss = q % 7;
                       v = ll[((size_t)ss * HEADS + hh) * BB + b]; }
    else             { q = f - 58; hh = q / 7; ss = q % 7;
                       v = ll[((size_t)(WALK + ss) * HEADS + hh) * BB + b]; }
    feat[f] = v;
  }
  __syncthreads();
  float a = b1[j];
  for (int f = 0; f < 72; ++f) a += feat[f] * W1[f * HH + j];
  a = a > 0.f ? a : 0.f;
  red[j] = a * W2[j];
  __syncthreads();
  if (j == 0) { float s = b2[0]; for (int t = 0; t < 32; ++t) s += red[t]; out[b] = s; }
}

// ---------- host ----------
extern "C" void kernel_launch(void* const* d_in, const int* in_sizes, int n_in,
                              void* d_out, int out_size, void* d_ws, size_t ws_size,
                              hipStream_t stream) {
  (void)in_sizes; (void)n_in; (void)out_size; (void)ws_size;
  const float* z_table = (const float*)d_in[0];
  const float* conv_W  = (const float*)d_in[1];
  const float* conv_b  = (const float*)d_in[2];
  const float* k1W = (const float*)d_in[3];  const float* k1b = (const float*)d_in[4];
  const float* q1W = (const float*)d_in[5];  const float* q1b = (const float*)d_in[6];
  const float* k2W = (const float*)d_in[7];  const float* k2b = (const float*)d_in[8];
  const float* q2W = (const float*)d_in[9];  const float* q2b = (const float*)d_in[10];
  const float* W1  = (const float*)d_in[11]; const float* b1  = (const float*)d_in[12];
  const float* W2  = (const float*)d_in[13]; const float* b2  = (const float*)d_in[14];
  const int* z    = (const int*)d_in[15];
  const int* erow = (const int*)d_in[16];
  const int* ecol = (const int*)d_in[17];
  // d_in[18]=edge_mask, d_in[19]=batch: derivable (e<E_REAL, n>>7) -> not read
  const int* node_i = (const int*)d_in[20];
  const int* node_j = (const int*)d_in[21];
  float* out = (float*)d_out;

  // workspace carve-out (~150 MB total)
  char* p = (char*)d_ws;
  auto alloc = [&](size_t bytes) { char* r = p; p += (bytes + 255) & ~(size_t)255; return r; };
  float* dinv  = (float*)alloc((size_t)NN * 4);
  float* xbuf  = (float*)alloc((size_t)NN * HH * 4);
  float* xwbuf = (float*)alloc((size_t)NN * HH * 4);
  float* agg   = (float*)alloc((size_t)NN * HH * 4);
  float* qa    = (float*)alloc((size_t)NN * HH * 4);
  float* ka    = (float*)alloc((size_t)NN * HH * 4);
  float* qb    = (float*)alloc((size_t)NN * 64 * 4);
  float* kb    = (float*)alloc((size_t)NN * 64 * 4);
  float* wbuf  = (float*)alloc((size_t)EE * HEADS * 4);
  float* ewbuf = (float*)alloc((size_t)EE * HEADS * 4);
  uint32_t* wkey = (uint32_t*)alloc((size_t)NN * HEADS * 4);
  float* sump  = (float*)alloc((size_t)NN * HEADS * 4);
  float* summ  = (float*)alloc((size_t)NN * HEADS * 4);
  float* dp    = (float*)alloc(DSZ * 4);
  float* dm    = (float*)alloc(DSZ * 4);
  unsigned short* dpb = (unsigned short*)alloc(DSZ * 2);
  unsigned short* dmb = (unsigned short*)alloc(DSZ * 2);
  float* trf = (float*)alloc((size_t)2 * WALK * HEADS * BB * 4);
  float* nlf = (float*)alloc((size_t)2 * WALK * HEADS * BB * 4);
  float* llf = (float*)alloc((size_t)2 * WALK * HEADS * BB * 4);
  float* omf = (float*)alloc((size_t)BB * HEADS * 4);

  const int TB = 256;
  auto nb = [](size_t n) { return (int)((n + 255) / 256); };

  // init
  k_fillf<<<nb(NN), TB, 0, stream>>>(dinv, NN, 1.0f);          // self-loop in degree
  k_fillu<<<nb((size_t)NN * HEADS), TB, 0, stream>>>(wkey, (size_t)NN * HEADS, 0u);
  k_fillf<<<nb((size_t)NN * HEADS), TB, 0, stream>>>(sump, (size_t)NN * HEADS, 0.f);
  k_fillf<<<nb((size_t)NN * HEADS), TB, 0, stream>>>(summ, (size_t)NN * HEADS, 0.f);
  k_fillf<<<nb(DSZ), TB, 0, stream>>>(dp, DSZ, 0.f);
  k_fillf<<<nb(DSZ), TB, 0, stream>>>(dm, DSZ, 0.f);

  // degrees, embedding
  k_deg<<<nb(EE), TB, 0, stream>>>(ecol, dinv);
  k_dinv<<<nb(NN), TB, 0, stream>>>(dinv);
  k_embed<<<nb((size_t)NN * HH), TB, 0, stream>>>(z_table, z, xbuf);

  // 3 GCN layers
  for (int l = 0; l < 3; ++l) {
    k_xw<<<nb((size_t)NN * HH), TB, 0, stream>>>(xbuf, conv_W + l * HH * HH, xwbuf);
    k_selfinit<<<nb((size_t)NN * HH), TB, 0, stream>>>(xwbuf, dinv, conv_b + l * HH, agg);
    k_scatter_conv<<<nb((size_t)EE * HH), TB, 0, stream>>>(erow, ecol, dinv, xwbuf, agg);
    k_act<<<nb((size_t)NN * HH), TB, 0, stream>>>(agg, xbuf, l < 2 ? 1 : 0);
  }

  // attention + per-edge softmax weights
  k_qk1<<<nb((size_t)NN * HH), TB, 0, stream>>>(xbuf, k1W, k1b, q1W, q1b, qa, ka);
  k_qk2<<<nb((size_t)NN * 64), TB, 0, stream>>>(qa, ka, k2W, k2b, q2W, q2b, qb, kb);
  k_edge_w<<<nb((size_t)EE * HEADS), TB, 0, stream>>>(erow, ecol, qb, kb, wbuf, wkey);
  k_edge_ew<<<nb((size_t)EE * HEADS), TB, 0, stream>>>(ecol, wbuf, wkey, ewbuf, sump, summ);
  k_omega<<<nb((size_t)BB * HEADS), TB, 0, stream>>>(wbuf, omf);

  // dense per-graph walk matrices -> bf16
  k_scatter_dense<<<nb((size_t)EE * HEADS), TB, 0, stream>>>(erow, ecol, ewbuf, sump, summ, dp, dm);
  k_tobf<<<nb(DSZ), TB, 0, stream>>>(dp, dpb, DSZ);
  k_tobf<<<nb(DSZ), TB, 0, stream>>>(dm, dmb, DSZ);

  // walk pooling: 1024 blocks (graph,head,variant), 96KB LDS each, WMMA chain
  k_walk<<<BB * HEADS * 2, 256, 3 * NP * NP * sizeof(unsigned short), stream>>>(
      dpb, dmb, node_i, node_j, trf, nlf, llf);

  // feature assembly + MLP
  k_mlp<<<BB, 32, 0, stream>>>(trf, nlf, llf, omf, W1, b1, W2, b2, out);
}
